// MoELayer_40707700032216
// MI455X (gfx1250) — matrix operand order repaired and checked by reading, not compiled, not run on hardware
//
#include <hip/hip_runtime.h>
#include <stdint.h>

// ---------------- problem constants ----------------
#define D_ 1024
#define H_ 4096
#define O_ 1024
#define E_ 8
#define T_ 4096          // B*S = 4*1024
#define HC 512           // H-chunk held in LDS between the two GEMMs
#define XA_STRIDE 1032   // 16x1024 bf16 X tile, +8 pad (2064B row -> bank-conflict free frag loads)
#define HS_STRIDE 520    // 16x512 bf16 gelu(h) tile, +8 pad
#define TILES_PER_E 256  // worst case: one expert owns all T tokens -> T/16 tiles

typedef __bf16 v16bf __attribute__((ext_vector_type(16)));
typedef float  v8f   __attribute__((ext_vector_type(8)));

union Frag {
  uint4    q[2];
  v16bf    v;
};

__device__ __forceinline__ uint16_t f2bf(float f) {
  uint32_t u = __float_as_uint(f);
  uint32_t r = u + 0x7FFFu + ((u >> 16) & 1u);  // round-to-nearest-even
  return (uint16_t)(r >> 16);
}
__device__ __forceinline__ float geluf(float x) {   // exact GELU (matches approximate=False)
  return 0.5f * x * (1.0f + erff(x * 0.70710678118654752440f));
}

// ---------------- 1) f32 -> bf16 transpose-convert: [e][R][C] -> [e][C][R] ----------------
__global__ void k_tconv(const float* __restrict__ src, uint16_t* __restrict__ dst,
                        int R, int C) {
  __shared__ float tile[32][33];
  int e = blockIdx.z;
  const float* s = src + (size_t)e * R * C;
  uint16_t*    d = dst + (size_t)e * R * C;
  int r0 = blockIdx.y * 32, c0 = blockIdx.x * 32;
#pragma unroll
  for (int j = 0; j < 4; ++j)
    tile[threadIdx.y + 8 * j][threadIdx.x] =
        s[(size_t)(r0 + threadIdx.y + 8 * j) * C + c0 + threadIdx.x];
  __syncthreads();
#pragma unroll
  for (int j = 0; j < 4; ++j)
    d[(size_t)(c0 + threadIdx.y + 8 * j) * R + r0 + threadIdx.x] =
        f2bf(tile[threadIdx.x][threadIdx.y + 8 * j]);
}

// ---------------- 2) gating: one wave per token; also emits bf16 X ----------------
__global__ void k_gate(const float* __restrict__ x, const float* __restrict__ Wg,
                       uint16_t* __restrict__ xbf, int* __restrict__ gate_e,
                       float* __restrict__ gate_w, int* __restrict__ cnt) {
  int wave = threadIdx.x >> 5, lane = threadIdx.x & 31;
  int t = blockIdx.x * 8 + wave;
  const float* xr = x + (size_t)t * D_;
  float acc[E_];
#pragma unroll
  for (int e = 0; e < E_; ++e) acc[e] = 0.f;
  for (int d = lane; d < D_; d += 32) {
    float xv = xr[d];
    xbf[(size_t)t * D_ + d] = f2bf(xv);
    const float4* wgr = (const float4*)(Wg + (size_t)d * E_);
    float4 a = wgr[0], b = wgr[1];
    acc[0] += xv * a.x; acc[1] += xv * a.y; acc[2] += xv * a.z; acc[3] += xv * a.w;
    acc[4] += xv * b.x; acc[5] += xv * b.y; acc[6] += xv * b.z; acc[7] += xv * b.w;
  }
#pragma unroll
  for (int e = 0; e < E_; ++e) {
#pragma unroll
    for (int off = 16; off >= 1; off >>= 1) acc[e] += __shfl_xor(acc[e], off, 32);
  }
  if (lane == 0) {
    int i0 = 0; float v0 = acc[0];
#pragma unroll
    for (int e = 1; e < E_; ++e) if (acc[e] > v0) { v0 = acc[e]; i0 = e; }
    int i1 = 0; float v1 = -3.4e38f;
#pragma unroll
    for (int e = 0; e < E_; ++e) if (e != i0 && acc[e] > v1) { v1 = acc[e]; i1 = e; }
    float ex = expf(v1 - v0);                 // softmax over the two selected logits
    float w0 = 1.f / (1.f + ex);
    gate_e[2 * t] = i0; gate_e[2 * t + 1] = i1;
    gate_w[2 * t] = w0; gate_w[2 * t + 1] = ex * w0;
    atomicAdd(&cnt[i0], 1);
    atomicAdd(&cnt[i1], 1);
  }
}

// ---------------- 3) routing: init / scan / scatter ----------------
__global__ void k_init(int* __restrict__ cnt, int* __restrict__ cursor) {
  int i = threadIdx.x;
  if (i < E_) { cnt[i] = 0; cursor[i] = 0; }
}
__global__ void k_scan(const int* __restrict__ cnt, int* __restrict__ offs) {
  if (threadIdx.x == 0) {
    int s = 0;
    for (int e = 0; e < E_; ++e) { offs[e] = s; s += cnt[e]; }
  }
}
__global__ void k_scatter(const int* __restrict__ gate_e, const float* __restrict__ gate_w,
                          const int* __restrict__ offs, int* __restrict__ cursor,
                          int* __restrict__ elist, float* __restrict__ ewt,
                          int* __restrict__ pos) {
  int t = blockIdx.x * 256 + threadIdx.x;
#pragma unroll
  for (int k = 0; k < 2; ++k) {
    int e = gate_e[2 * t + k];
    int slot = atomicAdd(&cursor[e], 1);          // integer only: FP result slot-order invariant
    int g = offs[e] + slot;
    elist[g] = t;
    ewt[g]   = gate_w[2 * t + k];
    pos[2 * t + k] = g;
  }
}

// ---------------- 4) fused expert FFN: 16 tokens/block, bf16 WMMA, GELU fused ----------------
__global__ void __launch_bounds__(256, 1) k_moe(
    const uint16_t* __restrict__ w1t,   // [E][H][D] bf16 (transposed)
    const uint16_t* __restrict__ w2t,   // [E][O][H] bf16 (transposed)
    const uint16_t* __restrict__ xbf,   // [T][D]    bf16
    const float* __restrict__ b1, const float* __restrict__ b2,
    const int* __restrict__ elist, const int* __restrict__ offs,
    const int* __restrict__ cnt, float* __restrict__ ybuf) {
  int e    = blockIdx.x / TILES_PER_E;
  int tile = blockIdx.x % TILES_PER_E;
  int n = cnt[e];
  if (tile * 16 >= n) return;                     // uniform per block
  int gbase = offs[e] + tile * 16;

  __shared__ uint16_t Xa[16 * XA_STRIDE];         // X tile, bf16
  __shared__ uint16_t Hs[16 * HS_STRIDE];         // gelu(h) chunk, bf16
  __shared__ int s_tok[16];
  __shared__ int s_val[16];

  int tid  = threadIdx.x;
  int wave = tid >> 5, lane = tid & 31;
  int half = lane >> 4, mlane = lane & 15;

  if (tid < 16) {
    int valid = (tile * 16 + tid) < n;
    s_tok[tid] = elist[valid ? (gbase + tid) : gbase];
    s_val[tid] = valid;
  }
  __syncthreads();

  // stage first 512 bf16 cols of 16 X rows (gathered by token id) into LDS
  for (int i = tid; i < 16 * 64; i += 256) {
    int r = i >> 6, c = i & 63;
    const uint4* src = (const uint4*)(xbf + (size_t)s_tok[r] * D_) + c;
    *(uint4*)(&Xa[r * XA_STRIDE + c * 8]) = *src;
  }
  __syncthreads();

  const v8f vzero = {0.f, 0.f, 0.f, 0.f, 0.f, 0.f, 0.f, 0.f};
  v8f yacc[8];                                    // this wave's 16x128 slice of Y
#pragma unroll
  for (int i = 0; i < 8; ++i) yacc[i] = vzero;

  for (int hc = 0; hc < H_; hc += HC) {
    // ---- GEMM1: gelu(X @ W1[:, hc..hc+HC) + b1); each wave owns 64 h-columns.
    //      kb-outer / nt-inner: one A-fragment (LDS) feeds 4 WMMAs.
    v8f acc1[4];
#pragma unroll
    for (int nt = 0; nt < 4; ++nt) acc1[nt] = vzero;
    const uint16_t* brow[4];
#pragma unroll
    for (int nt = 0; nt < 4; ++nt) {
      int hcol = hc + wave * 64 + nt * 16;
      brow[nt] = w1t + ((size_t)e * H_ + (hcol + mlane)) * D_;
    }
    for (int kb = 0; kb < D_ / 32; ++kb) {
      Frag a;
      const uint16_t* ap = &Xa[mlane * XA_STRIDE + kb * 32 + half * 8];
      a.q[0] = *(const uint4*)ap;                 // K = kb*32 + half*8 + 0..7
      a.q[1] = *(const uint4*)(ap + 16);          // K = +16
#pragma unroll
      for (int nt = 0; nt < 4; ++nt) {
        const uint16_t* bp = brow[nt] + kb * 32 + half * 8;
        Frag b;
        b.q[0] = *(const uint4*)bp;               // K = kb*32 + half*8 + 0..7
        b.q[1] = *(const uint4*)(bp + 16);        // K = +16
        acc1[nt] = __builtin_amdgcn_wmma_f32_16x16x32_bf16(
            false, a.v, false, b.v, (short)0, acc1[nt], false, false);
      }
    }
#pragma unroll
    for (int nt = 0; nt < 4; ++nt) {
      int hl   = wave * 64 + nt * 16;             // local column base inside chunk
      float b1v = b1[(size_t)e * H_ + hc + hl + mlane];
#pragma unroll
      for (int v = 0; v < 8; ++v) {
        int m = v + half * 8;                     // C/D layout: VGPR v <-> row m (+8 for hi lanes)
        Hs[m * HS_STRIDE + hl + mlane] = f2bf(geluf(acc1[nt][v] + b1v));
      }
    }
    __syncthreads();

    // ---- GEMM2: Y += gelu_h @ W2[hc..hc+HC, :); each wave owns 128 O-columns.
    //      kb-outer / nt-inner: one A-fragment (LDS) feeds 8 WMMAs.
    for (int kb = 0; kb < HC / 32; ++kb) {
      Frag a;
      const uint16_t* ap = &Hs[mlane * HS_STRIDE + kb * 32 + half * 8];
      a.q[0] = *(const uint4*)ap;
      a.q[1] = *(const uint4*)(ap + 16);
#pragma unroll
      for (int nt = 0; nt < 8; ++nt) {
        int o = wave * 128 + nt * 16 + mlane;
        const uint16_t* bp = w2t + ((size_t)e * O_ + o) * H_ + hc + kb * 32 + half * 8;
        Frag b;
        b.q[0] = *(const uint4*)bp;
        b.q[1] = *(const uint4*)(bp + 16);
        yacc[nt] = __builtin_amdgcn_wmma_f32_16x16x32_bf16(
            false, a.v, false, b.v, (short)0, yacc[nt], false, false);
      }
    }
    __syncthreads();
  }

  // epilogue: ybuf[gbase+m][o] = Y + b2  (unweighted; gate weight applied at combine)
#pragma unroll
  for (int nt = 0; nt < 8; ++nt) {
    int o = wave * 128 + nt * 16 + mlane;
    float b2v = b2[(size_t)e * O_ + o];
#pragma unroll
    for (int v = 0; v < 8; ++v) {
      int m = v + half * 8;
      if (s_val[m])
        ybuf[(size_t)(gbase + m) * O_ + o] = yacc[nt][v] + b2v;
    }
  }
}

// ---------------- 5) deterministic combine: out = w0*y(slot0) + w1*y(slot1) ----------------
__global__ void k_combine(const float* __restrict__ ybuf, const int* __restrict__ pos,
                          const float* __restrict__ ewt, float* __restrict__ out) {
  int gid = blockIdx.x * 256 + threadIdx.x;       // T*O elements
  int t = gid >> 10, o = gid & (O_ - 1);
  int p0 = pos[2 * t], p1 = pos[2 * t + 1];
  out[gid] = ewt[p0] * ybuf[(size_t)p0 * O_ + o] +
             ewt[p1] * ybuf[(size_t)p1 * O_ + o];
}

// ---------------- launch ----------------
extern "C" void kernel_launch(void* const* d_in, const int* in_sizes, int n_in,
                              void* d_out, int out_size, void* d_ws, size_t ws_size,
                              hipStream_t stream) {
  (void)in_sizes; (void)n_in; (void)out_size; (void)ws_size;
  const float* x  = (const float*)d_in[0];
  const float* Wg = (const float*)d_in[1];
  const float* W1 = (const float*)d_in[2];
  const float* b1 = (const float*)d_in[3];
  const float* W2 = (const float*)d_in[4];
  const float* b2 = (const float*)d_in[5];
  float* out = (float*)d_out;

  char* base = (char*)d_ws;
  size_t off = 0;
  auto take = [&](size_t bytes) -> void* {
    void* p = base + off;
    off += (bytes + 255) & ~(size_t)255;
    return p;
  };
  uint16_t* w1t  = (uint16_t*)take((size_t)E_ * D_ * H_ * 2);   //  64 MB
  uint16_t* w2t  = (uint16_t*)take((size_t)E_ * H_ * O_ * 2);   //  64 MB
  uint16_t* xbf  = (uint16_t*)take((size_t)T_ * D_ * 2);        //   8 MB
  float*    ybuf = (float*)take((size_t)2 * T_ * O_ * 4);       //  32 MB
  int*   gate_e  = (int*)take((size_t)2 * T_ * 4);
  float* gate_w  = (float*)take((size_t)2 * T_ * 4);
  int*   elist   = (int*)take((size_t)2 * T_ * 4);
  float* ewt     = (float*)take((size_t)2 * T_ * 4);
  int*   pos     = (int*)take((size_t)2 * T_ * 4);
  int*   cnt     = (int*)take(E_ * 4);
  int*   offs    = (int*)take(E_ * 4);
  int*   cursor  = (int*)take(E_ * 4);

  k_init<<<1, 32, 0, stream>>>(cnt, cursor);
  k_tconv<<<dim3(H_ / 32, D_ / 32, E_), dim3(32, 8), 0, stream>>>(W1, w1t, D_, H_);
  k_tconv<<<dim3(O_ / 32, H_ / 32, E_), dim3(32, 8), 0, stream>>>(W2, w2t, H_, O_);
  k_gate<<<T_ / 8, 256, 0, stream>>>(x, Wg, xbf, gate_e, gate_w, cnt);
  k_scan<<<1, 1, 0, stream>>>(cnt, offs);
  k_scatter<<<T_ / 256, 256, 0, stream>>>(gate_e, gate_w, offs, cursor, elist, ewt, pos);
  k_moe<<<E_ * TILES_PER_E, 256, 0, stream>>>(w1t, w2t, xbf, b1, b2, elist, offs, cnt, ybuf);
  k_combine<<<(T_ * O_) / 256, 256, 0, stream>>>(ybuf, pos, ewt, out);
}
